// BackboneGRUD_42004780155214
// MI455X (gfx1250) — matrix operand order repaired and checked by reading, not compiled, not run on hardware
//
#include <hip/hip_runtime.h>
#include <hip/hip_bf16.h>

typedef __attribute__((ext_vector_type(16))) _Float16 v16h;
typedef __attribute__((ext_vector_type(4)))  _Float16 v4h;
typedef __attribute__((ext_vector_type(8)))  float    v8f;
typedef __attribute__((ext_vector_type(4)))  float    v4f;

// Problem dims
constexpr int Bsz  = 1024;
constexpr int Tlen = 128;
constexpr int Fdim = 128;   // features
constexpr int Hdim = 256;   // hidden
constexpr int KI   = 2*Fdim + Hdim;   // 512  (concat input width)
constexpr int NG   = 3*Hdim;          // 768  (gate output width)

// LDS row strides (halves), padded to dodge bank conflicts on A-frag reads
constexpr int INP_LD = KI + 8;        // 520 halves (row = 1040 B, 8B aligned)
constexpr int D_LD   = Fdim + 8;      // 136 halves (row = 272 B, 8B aligned)

// Packed-weight fragment counts: [N/16][K/32][32 lanes][16 halves]
constexpr int WIH_HALVES = (NG/16)*(KI/32)*32*16;    // 393216
constexpr int WHH_HALVES = (NG/16)*(Hdim/32)*32*16;  // 196608
constexpr int WDH_HALVES = (Hdim/16)*(Fdim/32)*32*16;// 32768

// Shared memory budget (bytes)
constexpr size_t SMEM_BYTES =
    (size_t)(5*16*Hdim + 16*Fdim + Fdim + Fdim) * 4   // h,r,z,in,hn,mu,diag,bdx (f32)
  + (size_t)(16*INP_LD + 16*D_LD) * 2;                // inp, d (f16)

// ---------------------------------------------------------------------------
// Pack a row-major f32 weight W[Nmat][Kmat] into per-lane WMMA B-fragment
// order (f16): dst laid out as [nt][kt][lane][8 dwords]; lane<16 holds column
// n = nt*16+lane with K = kt*32 + 2v; lane>=16 holds K = kt*32 + 16 + 2v.
// B[k][n] = W[n][k]   (we compute inp @ W^T).
// ---------------------------------------------------------------------------
__global__ void pack_wmma_b(const float* __restrict__ W,
                            _Float16* __restrict__ dst,
                            int Nmat, int Kmat) {
  int i = blockIdx.x * blockDim.x + threadIdx.x;
  int KT = Kmat >> 5;
  int total = (Nmat >> 4) * KT * 32 * 8;   // one thread per dword (2 halves)
  if (i >= total) return;
  int v    = i & 7;
  int lane = (i >> 3) & 31;
  int tile = i >> 8;            // nt*KT + kt
  int kt   = tile % KT;
  int nt   = tile / KT;
  int n = (nt << 4) + (lane & 15);
  int k = (kt << 5) + ((lane & 16) ? 16 : 0) + (v << 1);
  dst[(size_t)i*2 + 0] = (_Float16)W[(size_t)n * Kmat + k];
  dst[(size_t)i*2 + 1] = (_Float16)W[(size_t)n * Kmat + k + 1];
}

// ---------------------------------------------------------------------------
// Load a 16-bit A fragment (16-row tile, K-window of 32 at kbase) from an LDS
// row buffer, in contiguous k order (same as pack_wmma_b):
//   lanes 0-15 : M = lane, VGPR v -> K = kbase+2v
//   lanes 16-31: same M, K = kbase+16+2v.
// ---------------------------------------------------------------------------
__device__ __forceinline__ v16h load_a_frag(const _Float16* rowbase, int kbase, int lane) {
  const unsigned int* r = (const unsigned int*)rowbase;   // 2 halves per dword
  int b0 = (kbase >> 1) + ((lane & 16) ? 8 : 0);
  union { v16h v; unsigned int u[8]; } a;
#pragma unroll
  for (int q = 0; q < 4; ++q) a.u[q]     = r[b0 + q];
#pragma unroll
  for (int q = 0; q < 4; ++q) a.u[4 + q] = r[b0 + 4 + q];
  return a.v;
}

__device__ __forceinline__ v8f wmma_f16(v16h a, v16h b, v8f c) {
  return __builtin_amdgcn_wmma_f32_16x16x32_f16(false, a, false, b, (short)0, c, false, false);
}

__device__ __forceinline__ float sigmoidf_fast(float x) {
  return 1.0f / (1.0f + __expf(-x));
}

// ---------------------------------------------------------------------------
// Persistent GRU-D scan kernel: 64 blocks x 256 threads (8 wave32 waves).
// Each block owns a 16-row batch tile and iterates all T=128 steps.
// Activation streaming (the only per-step HBM reads) is software-pipelined
// one step ahead in registers so its latency is hidden under the WMMA phases.
// ---------------------------------------------------------------------------
__global__ __launch_bounds__(256) void grud_scan_kernel(
    const float* __restrict__ X,  const float* __restrict__ Mk,
    const float* __restrict__ Dl, const float* __restrict__ mu,
    const float* __restrict__ Xl, const float* __restrict__ Wdx,
    const float* __restrict__ bdx, const float* __restrict__ bdh,
    const float* __restrict__ bih, const float* __restrict__ bhh,
    const _Float16* __restrict__ pWih, const _Float16* __restrict__ pWhh,
    const _Float16* __restrict__ pWdh,
    float* __restrict__ reps, float* __restrict__ hfin) {

  extern __shared__ char smem_raw[];
  float* h_s    = (float*)smem_raw;          // 16 x 256
  float* r_s    = h_s  + 16*Hdim;            // 16 x 256
  float* z_s    = r_s  + 16*Hdim;            // 16 x 256
  float* in_s   = z_s  + 16*Hdim;            // 16 x 256 (i_n part)
  float* hn_s   = in_s + 16*Hdim;            // 16 x 256 (h_n part)
  float* mu_s   = hn_s + 16*Hdim;            // 16 x 128
  float* diag_s = mu_s + 16*Fdim;            // 128
  float* bdx_s  = diag_s + Fdim;             // 128
  _Float16* inp_s = (_Float16*)(bdx_s + Fdim);   // 16 x INP_LD
  _Float16* d_s   = inp_s + 16*INP_LD;           // 16 x D_LD

  const int tid   = threadIdx.x;
  const int lane  = tid & 31;
  const int wave  = tid >> 5;
  const int m0    = blockIdx.x * 16;     // first batch row of this tile
  const int mrow  = lane & 15;           // A-frag row for this lane
  const int Mbase = (lane >> 4) << 3;    // C-frag row base (0 or 8)

  const v16h* pWdh_v = (const v16h*)pWdh;   // small: OK to keep resident

  // ---- one-time staging: h0 = 0, cache mu row-tile, diag(Wdx), bdx --------
  for (int i = tid; i < 16*Hdim; i += 256) h_s[i] = 0.0f;
  for (int i = tid; i < 16*Fdim; i += 256)
    mu_s[i] = mu[(size_t)(m0 + (i >> 7)) * Fdim + (i & 127)];
  for (int i = tid; i < Fdim; i += 256) {
    diag_s[i] = Wdx[(size_t)i * Fdim + i];
    bdx_s[i]  = bdx[i];
  }
  __syncthreads();

  const int Ng0 = (wave*6)*16 + (lane & 15);     // base N for this wave's tiles

  // ---- activation stream pipeline: each thread owns 2 float4 quads/step ---
  int eMr[2], ef[2];
#pragma unroll
  for (int s = 0; s < 2; ++s) {
    int e = tid + s*256;
    eMr[s] = e >> 5;            // Fdim/4 == 32 quads per row
    ef[s]  = (e & 31) << 2;
  }
  v4f pd[2], pxl[2], pm[2], px[2];
#pragma unroll
  for (int s = 0; s < 2; ++s) {      // prefetch t = 0
    size_t idx4 = (((size_t)(m0 + eMr[s]) * Tlen + 0) * Fdim + ef[s]) >> 2;
    pd[s]  = ((const v4f*)Dl)[idx4];
    pxl[s] = ((const v4f*)Xl)[idx4];
    pm[s]  = ((const v4f*)Mk)[idx4];
    px[s]  = ((const v4f*)X )[idx4];
  }

  for (int t = 0; t < Tlen; ++t) {
    // Defeat LICM on the big weight arrays: fragments must be re-fetched from
    // L0/L2 each step instead of being hoisted (1216 VGPRs) and spilled.
    const v16h* wih_v = (const v16h*)pWih;
    const v16h* whh_v = (const v16h*)pWhh;
    asm volatile("" : "+s"(wih_v), "+s"(whh_v));

    // ---- Phase 1: consume prefetched quads -> gamma_x / x_in -> LDS -------
#pragma unroll
    for (int s = 0; s < 2; ++s) {
      int Mr = eMr[s], f = ef[s];
      v4h dh, xinh, mh;
#pragma unroll
      for (int q = 0; q < 4; ++q) {
        float d  = pd[s][q];
        float gx = __expf(-fmaxf(fmaf(d, diag_s[f+q], bdx_s[f+q]), 0.0f));
        float xh = gx * pxl[s][q] + (1.0f - gx) * mu_s[Mr*Fdim + f + q];
        float mm = pm[s][q];
        float xin = mm * px[s][q] + (1.0f - mm) * xh;
        dh[q]   = (_Float16)d;
        xinh[q] = (_Float16)xin;
        mh[q]   = (_Float16)mm;
      }
      *(v4h*)&d_s[Mr*D_LD + f]                    = dh;
      *(v4h*)&inp_s[Mr*INP_LD + f]                = xinh;   // cols [0,F)
      *(v4h*)&inp_s[Mr*INP_LD + Fdim + Hdim + f]  = mh;     // cols [F+H,..)
    }
    // Issue next step's loads now; they complete under phases 2-4.
    {
      int tn = (t + 1 < Tlen) ? (t + 1) : (Tlen - 1);
#pragma unroll
      for (int s = 0; s < 2; ++s) {
        size_t idx4 = (((size_t)(m0 + eMr[s]) * Tlen + tn) * Fdim + ef[s]) >> 2;
        pd[s]  = ((const v4f*)Dl)[idx4];
        pxl[s] = ((const v4f*)Xl)[idx4];
        pm[s]  = ((const v4f*)Mk)[idx4];
        px[s]  = ((const v4f*)X )[idx4];
      }
    }
    __syncthreads();

    // ---- Phase 2: gamma_h = exp(-relu(d @ Wdh^T + bdh)); h *= gamma_h -----
    // 16 N-tiles of H=256; 2 per wave; K = 128 -> 4 WMMA each.
    {
      v16h afrag[4];
#pragma unroll
      for (int kt = 0; kt < 4; ++kt)
        afrag[kt] = load_a_frag(d_s + mrow*D_LD, kt*32, lane);
#pragma unroll
      for (int tt = 0; tt < 2; ++tt) {
        int nt = wave*2 + tt;
        v8f acc = {};
#pragma unroll
        for (int kt = 0; kt < 4; ++kt) {
          v16h b = pWdh_v[(size_t)(nt*4 + kt)*32 + lane];
          acc = wmma_f16(afrag[kt], b, acc);
        }
        int N = nt*16 + (lane & 15);
        float bias = bdh[N];
#pragma unroll
        for (int j = 0; j < 8; ++j) {
          int Mr = Mbase + j;
          float g  = __expf(-fmaxf(acc[j] + bias, 0.0f));
          float hv = h_s[Mr*Hdim + N] * g;
          h_s[Mr*Hdim + N] = hv;                           // decayed state
          inp_s[Mr*INP_LD + Fdim + N] = (_Float16)hv;      // cols [F, F+H)
          reps[((size_t)(m0 + Mr) * Tlen + t) * Hdim + N] = hv;  // rep output
        }
      }
    }
    __syncthreads();

    // ---- Phase 3: gi = inp @ Wih^T (+bih), gh = h @ Whh^T (+bhh) ----------
    // 48 N-tiles of 768; 6 per wave. gi: K=512 (16 kt), gh: K=256 (8 kt).
    {
      v8f accg[6] = {};
      v8f acch[6] = {};
      for (int kt = 0; kt < 16; ++kt) {
        v16h a = load_a_frag(inp_s + mrow*INP_LD, kt*32, lane);
#pragma unroll
        for (int nt = 0; nt < 6; ++nt) {
          v16h b = wih_v[((size_t)(wave*6 + nt)*16 + kt)*32 + lane];
          accg[nt] = wmma_f16(a, b, accg[nt]);
        }
      }
      for (int kt = 0; kt < 8; ++kt) {
        v16h a = load_a_frag(inp_s + mrow*INP_LD, Fdim + kt*32, lane);
#pragma unroll
        for (int nt = 0; nt < 6; ++nt) {
          v16h b = whh_v[((size_t)(wave*6 + nt)*8 + kt)*32 + lane];
          acch[nt] = wmma_f16(a, b, acch[nt]);
        }
      }
      // Gate post-processing: r/z -> sigmoid now; n parts exchanged via LDS.
#pragma unroll
      for (int nt = 0; nt < 6; ++nt) {
        int N  = Ng0 + nt*16;               // 0..767
        float bg = bih[N], bh = bhh[N];
        int gate = N >> 8;                  // 0=r, 1=z, 2=n (sections of 256)
        int col  = N & 255;
#pragma unroll
        for (int j = 0; j < 8; ++j) {
          int Mr = Mbase + j;
          float vg = accg[nt][j] + bg;
          float vh = acch[nt][j] + bh;
          if (gate == 0) {
            r_s[Mr*Hdim + col] = sigmoidf_fast(vg + vh);
          } else if (gate == 1) {
            z_s[Mr*Hdim + col] = sigmoidf_fast(vg + vh);
          } else {
            in_s[Mr*Hdim + col] = vg;
            hn_s[Mr*Hdim + col] = vh;
          }
        }
      }
    }
    __syncthreads();

    // ---- Phase 4: h_new = (1-z)*tanh(i_n + r*h_n) + z*h -------------------
    for (int e = tid; e < 16*Hdim; e += 256) {
      float r = r_s[e], z = z_s[e];
      float n = tanhf(in_s[e] + r * hn_s[e]);
      float h = h_s[e];
      h_s[e] = (1.0f - z) * n + z * h;
    }
    __syncthreads();
  }

  // ---- final hidden state -------------------------------------------------
  for (int e = tid; e < 16*Hdim; e += 256) {
    int Mr = e >> 8, N = e & 255;
    hfin[(size_t)(m0 + Mr) * Hdim + N] = h_s[e];
  }
}

// ---------------------------------------------------------------------------
extern "C" void kernel_launch(void* const* d_in, const int* in_sizes, int n_in,
                              void* d_out, int out_size, void* d_ws, size_t ws_size,
                              hipStream_t stream) {
  const float* X    = (const float*)d_in[0];
  const float* Mk   = (const float*)d_in[1];
  const float* Dl   = (const float*)d_in[2];
  const float* mu   = (const float*)d_in[3];
  const float* Xl   = (const float*)d_in[4];
  const float* Wdh  = (const float*)d_in[5];
  const float* bdh  = (const float*)d_in[6];
  const float* Wdx  = (const float*)d_in[7];
  const float* bdx  = (const float*)d_in[8];
  const float* Wih  = (const float*)d_in[9];
  const float* Whh  = (const float*)d_in[10];
  const float* bih  = (const float*)d_in[11];
  const float* bhh  = (const float*)d_in[12];

  // Workspace: packed f16 WMMA B fragments (L2-resident in the hot loop).
  _Float16* pWih = (_Float16*)d_ws;
  _Float16* pWhh = pWih + WIH_HALVES;
  _Float16* pWdh = pWhh + WHH_HALVES;

  pack_wmma_b<<<(WIH_HALVES/2 + 255)/256, 256, 0, stream>>>(Wih, pWih, NG,   KI);
  pack_wmma_b<<<(WHH_HALVES/2 + 255)/256, 256, 0, stream>>>(Whh, pWhh, NG,   Hdim);
  pack_wmma_b<<<(WDH_HALVES/2 + 255)/256, 256, 0, stream>>>(Wdh, pWdh, Hdim, Fdim);

  float* reps = (float*)d_out;
  float* hfin = reps + (size_t)Bsz * Tlen * Hdim;

  grud_scan_kernel<<<Bsz/16, 256, SMEM_BYTES, stream>>>(
      X, Mk, Dl, mu, Xl, Wdx, bdx, bdh, bih, bhh,
      pWih, pWhh, pWdh, reps, hfin);
}
